// EGCL_63161789055082
// MI455X (gfx1250) — matrix operand order repaired and checked by reading, not compiled, not run on hardware
//
#include <hip/hip_runtime.h>
#include <hip/hip_bf16.h>

typedef __attribute__((ext_vector_type(16))) _Float16 v16h;
typedef __attribute__((ext_vector_type(8)))  _Float16 v8h;
typedef __attribute__((ext_vector_type(4)))  _Float16 v4h;
typedef __attribute__((ext_vector_type(8)))  float    v8f;

#define NNODES 512
#define NEDGE  511
#define NTILES 32            // ceil(511/16)

// ---- shared memory layout (bytes, all 16B aligned) ----
#define SM_WE0T  0                          // [128][160] f16 (We0^T, K padded 136->160)
#define SM_WE1T  (SM_WE0T + 128*160*2)      // [128][128] f16
#define SM_WX0T  (SM_WE1T + 128*128*2)
#define SM_WX1T  (SM_WX0T + 128*128*2)
#define SM_WTPT  (SM_WX1T + 128*128*2)      // [64][128] f16 (W_tp^T: [v*8+w][u])
#define SM_WINFT (SM_WTPT + 64*128*2)       // [16][128] f16 (w_inf in col 0, rest 0)
#define SM_MACC  (SM_WINFT + 16*128*2)      // [128] f32 block accumulator for m_i
#define SM_VACC  (SM_MACC + 512)            // [24]  f32 block accumulator for vectors_agg
#define SM_WAVE0 (SM_VACC + 128)
#define WAVE_BYTES 14912
#define SMEM_TOTAL (SM_WAVE0 + 8*WAVE_BYTES)   // = 279,680 B  (< 320KB WGP LDS)
// wave-local offsets
#define WL_BUFA 0        // [16][160] f16 edge_feat / [16][128] f16 h3 / [16][64] f32 g
#define WL_BUFB 5120     // [16][128] f16 h1 / phi
#define WL_BUFC 9216     // [16][128] f16 m_ij
#define WL_SH1  13312    // [16][24] f32 sqrt(3)*unit vectors
#define WL_GATE 14848    // [16] f32 sigmoid gates

// A fragment (16x32 f16): lanes 0-15 hold K = kc*32 + {0..7,16..23} of row (lane&15);
// lanes 16-31 hold K = kc*32 + {8..15,24..31}.
__device__ __forceinline__ v16h load_A(const _Float16* __restrict__ Abuf, int astride,
                                       int kc, int half_id, int row)
{
    const _Float16* pa = Abuf + row * astride + kc * 32 + (half_id << 3);
    v8h a0 = *(const v8h*)pa;
    v8h a1 = *(const v8h*)(pa + 16);
    v16h a;
#pragma unroll
    for (int t = 0; t < 8; ++t) { a[t] = a0[t]; a[8 + t] = a1[t]; }
    return a;
}

// B fragment (32x16 f16), weights stored transposed WT[n][k]:
// lanes 0-15: K = kc*32 + {0..7,16..23} of col n = nc*16 + (lane&15); lanes 16-31: K = kc*32 + {8..15,24..31}.
__device__ __forceinline__ v16h load_B(const _Float16* __restrict__ WT, int kstride,
                                       int nc, int kc, int half_id, int row)
{
    const _Float16* pb = WT + ((nc << 4) + row) * kstride + kc * 32 + (half_id << 3);
    v8h b0 = *(const v8h*)pb;
    v8h b1 = *(const v8h*)(pb + 16);
    v16h b;
#pragma unroll
    for (int t = 0; t < 8; ++t) { b[t] = b0[t]; b[8 + t] = b1[t]; }
    return b;
}

// Software-pipelined GEMM over one 16-edge tile:
// C[16, nch*16] = act( (A[16, kch*32] @ B) * scale ).  B = WT transposed in LDS.
__device__ __forceinline__ void gemm_tile(
    const _Float16* __restrict__ Abuf, int astride,
    const _Float16* __restrict__ WT,   int kstride,
    int kch, int nch,
    _Float16* Ch, float* Cf, int cstride,
    float scale, bool do_silu, int lane)
{
    const int half_id = lane >> 4;
    const int row     = lane & 15;
    v8f acc[8];
    const v8f zero = {0.f,0.f,0.f,0.f,0.f,0.f,0.f,0.f};
    for (int nc = 0; nc < nch; ++nc) acc[nc] = zero;

    // prime the pipeline
    v16h a = load_A(Abuf, astride, 0, half_id, row);
    v16h b = load_B(WT, kstride, 0, 0, half_id, row);

    for (int kc = 0; kc < kch; ++kc) {
        v16h anext = a;
        for (int nc = 0; nc < nch; ++nc) {
            v16h bnext = b;
            if (nc + 1 < nch) {
                bnext = load_B(WT, kstride, nc + 1, kc, half_id, row);     // prefetch next B
            } else if (kc + 1 < kch) {
                anext = load_A(Abuf, astride, kc + 1, half_id, row);       // prefetch next A
                bnext = load_B(WT, kstride, 0, kc + 1, half_id, row);
            }
            acc[nc] = __builtin_amdgcn_wmma_f32_16x16x32_f16(
                false, a, false, b, (short)0, acc[nc], false, false);
            b = bnext;
        }
        a = anext;
    }
    // C layout: element r lives at (M = half_id*8 + r, N = nc*16 + row)
    for (int nc = 0; nc < nch; ++nc) {
#pragma unroll
        for (int r = 0; r < 8; ++r) {
            float x = acc[nc][r] * scale;
            if (do_silu) x = x / (1.0f + __expf(-x));
            const int m = (half_id << 3) + r;
            const int n = (nc << 4) + row;
            if (Ch) Ch[m * cstride + n] = (_Float16)x;
            else    Cf[m * cstride + n] = x;
        }
    }
}

__global__ __launch_bounds__(256, 1)
void egcl_fused(const float* __restrict__ pos,     // [512][8][3]
                const float* __restrict__ feat,    // [512][64]
                const int*   __restrict__ senders, // [512][511]
                const float* __restrict__ We0,     // [136][128]
                const float* __restrict__ We1,     // [128][128]
                const float* __restrict__ winf,    // [128][1]
                const float* __restrict__ Wx0,     // [128][128]
                const float* __restrict__ Wx1,     // [128][128]
                const float* __restrict__ Wtp,     // [128][8][8]
                float* __restrict__ out_pos,       // [512][8][3]
                float* __restrict__ out_mi)        // [512][128]
{
    extern __shared__ char smem[];
    const int i    = blockIdx.x;
    const int tid  = threadIdx.x;
    const int wave = tid >> 5;
    const int lane = tid & 31;
    const int half_id = lane >> 4;
    const int row     = lane & 15;

    _Float16* We0T  = (_Float16*)(smem + SM_WE0T);
    _Float16* We1T  = (_Float16*)(smem + SM_WE1T);
    _Float16* Wx0T  = (_Float16*)(smem + SM_WX0T);
    _Float16* Wx1T  = (_Float16*)(smem + SM_WX1T);
    _Float16* WtpT  = (_Float16*)(smem + SM_WTPT);
    _Float16* WinfT = (_Float16*)(smem + SM_WINFT);
    float* maccs = (float*)(smem + SM_MACC);
    float* vaccs = (float*)(smem + SM_VACC);

    char* wbase = smem + SM_WAVE0 + wave * WAVE_BYTES;
    _Float16* bufA  = (_Float16*)(wbase + WL_BUFA);   // edge_feat [16][160] / h3 [16][128]
    _Float16* bufB  = (_Float16*)(wbase + WL_BUFB);   // h1 / phi  [16][128]
    _Float16* bufC  = (_Float16*)(wbase + WL_BUFC);   // m_ij      [16][128]
    float*    sh1b  = (float*)  (wbase + WL_SH1);     // [16][24]
    float*    gateb = (float*)  (wbase + WL_GATE);    // [16]
    float*    gbuf  = (float*)  (wbase + WL_BUFA);    // g [16][64] f32 (aliases bufA)

    // ---- stage weights: f32 global -> f16 transposed LDS (K-contiguous per output col) ----
    for (int idx = tid; idx < 128 * 160; idx += 256) {
        int n = idx / 160, k = idx - n * 160;
        We0T[idx] = (_Float16)((k < 136) ? We0[k * 128 + n] : 0.0f);
    }
    for (int idx = tid; idx < 128 * 128; idx += 256) {
        int n = idx >> 7, k = idx & 127;
        We1T[idx] = (_Float16)We1[k * 128 + n];
        Wx0T[idx] = (_Float16)Wx0[k * 128 + n];
        Wx1T[idx] = (_Float16)Wx1[k * 128 + n];
    }
    for (int idx = tid; idx < 64 * 128; idx += 256) {
        int n = idx >> 7, u = idx & 127;
        WtpT[idx] = (_Float16)Wtp[u * 64 + n];       // W_tp[u][v][w] -> WtpT[v*8+w][u]
    }
    for (int idx = tid; idx < 16 * 128; idx += 256) {
        int n = idx >> 7, k = idx & 127;
        WinfT[idx] = (_Float16)((n == 0) ? winf[k] : 0.0f);
    }
    if (tid < 128) maccs[tid] = 0.0f;
    if (tid < 24)  vaccs[tid] = 0.0f;
    __syncthreads();

    const float INV_S136 = 0.08574929257f;  // 1/sqrt(136)
    const float INV_S128 = 0.08838834765f;  // 1/sqrt(128)
    const float SQRT3    = 1.73205080757f;

    float macc[4] = {0.f, 0.f, 0.f, 0.f};   // per-lane partial of m_i, dims d = lane*4+q
    float vacc[12];                          // per-lane partials of tensor-product output
#pragma unroll
    for (int t = 0; t < 12; ++t) vacc[t] = 0.f;

    for (int tile = wave; tile < NTILES; tile += 8) {
        const int ebase = tile << 4;

        // ---- build edge features: [lengths(8) | feat_s(64) | feat_r(64) | pad(24)] f16 ----
#pragma unroll
        for (int it = 0; it < 4; ++it) {                 // 128 (edge, v) tasks
            int id = lane + 32 * it;
            int el = id >> 3, v = id & 7;
            int e = ebase + el;
            float dx = 0.f, dy = 0.f, dz = 0.f, len = 0.f, inv = 0.f;
            if (e < NEDGE) {
                int s = senders[i * NEDGE + e];
                dx = pos[s * 24 + v * 3 + 0] - pos[i * 24 + v * 3 + 0];
                dy = pos[s * 24 + v * 3 + 1] - pos[i * 24 + v * 3 + 1];
                dz = pos[s * 24 + v * 3 + 2] - pos[i * 24 + v * 3 + 2];
                float d2 = dx * dx + dy * dy + dz * dz;
                len = sqrtf(fmaxf(d2, 1e-12f));
                inv = SQRT3 / len;
            }
            bufA[el * 160 + v] = (_Float16)len;
            sh1b[el * 24 + v * 3 + 0] = dx * inv;
            sh1b[el * 24 + v * 3 + 1] = dy * inv;
            sh1b[el * 24 + v * 3 + 2] = dz * inv;
        }
#pragma unroll 4
        for (int it = 0; it < 32; ++it) {                // 1024 (edge, f) tasks
            int id = lane + 32 * it;
            int el = id >> 6, f = id & 63;
            int e = ebase + el;
            float vs = 0.f, vr = 0.f;
            if (e < NEDGE) {
                int s = senders[i * NEDGE + e];
                vs = feat[s * 64 + f];
                vr = feat[i * 64 + f];
            }
            bufA[el * 160 + 8 + f]  = (_Float16)vs;
            bufA[el * 160 + 72 + f] = (_Float16)vr;
        }
#pragma unroll
        for (int it = 0; it < 12; ++it) {                // zero K-pad cols 136..159
            int id = lane + 32 * it;
            int el = id / 24, k = 136 + id % 24;
            bufA[el * 160 + k] = (_Float16)0.0f;
        }

        // ---- edge MLP ----
        gemm_tile(bufA, 160, We0T, 160, 5, 8, bufB, nullptr, 128, INV_S136, true, lane); // h1
        gemm_tile(bufB, 128, We1T, 128, 4, 8, bufC, nullptr, 128, INV_S128, true, lane); // m_ij

        // ---- gate: dots = m_ij @ w_inf_tile via WMMA (only column 0 of the tile is live) ----
        {
            v8f acc = {0.f,0.f,0.f,0.f,0.f,0.f,0.f,0.f};
            v16h a = load_A(bufC, 128, 0, half_id, row);
            v16h b = load_B(WinfT, 128, 0, 0, half_id, row);
            for (int kc = 0; kc < 4; ++kc) {
                v16h an = a, bn = b;
                if (kc + 1 < 4) {
                    an = load_A(bufC, 128, kc + 1, half_id, row);
                    bn = load_B(WinfT, 128, 0, kc + 1, half_id, row);
                }
                acc = __builtin_amdgcn_wmma_f32_16x16x32_f16(
                    false, a, false, b, (short)0, acc, false, false);
                a = an; b = bn;
            }
            if (row == 0) {   // lanes 0 and 16 hold column N=0: dot for edge m = half_id*8+r
#pragma unroll
                for (int r = 0; r < 8; ++r) {
                    float x = acc[r] * INV_S128;
                    gateb[(half_id << 3) + r] = 1.0f / (1.0f + __expf(-x));
                }
            }
        }

        // ---- gated scatter-sum: macc[q] += sum_e m_ij[e][lane*4+q]*gate[e] ----
#pragma unroll
        for (int e2 = 0; e2 < 16; ++e2) {
            v4h mv = *(const v4h*)(bufC + e2 * 128 + lane * 4);
            float g = gateb[e2];
            macc[0] += (float)mv[0] * g;
            macc[1] += (float)mv[1] * g;
            macc[2] += (float)mv[2] * g;
            macc[3] += (float)mv[3] * g;
        }

        // ---- phi_x MLP ----
        gemm_tile(bufC, 128, Wx0T, 128, 4, 8, bufA, nullptr, 128, INV_S128, true, lane); // h3
        gemm_tile(bufA, 128, Wx1T, 128, 4, 8, bufB, nullptr, 128, INV_S128, true, lane); // phi

        // ---- tensor product: g[e][v*8+w] = phi[e] @ W_tp[:, v, w] ----
        gemm_tile(bufB, 128, WtpT, 128, 4, 4, nullptr, gbuf, 64, 1.0f, false, lane);

        // ---- vacc += sum_v sh1[e][v][c] * g[e][v*8+w]  (384 (e,w,c) tasks) ----
#pragma unroll
        for (int it = 0; it < 12; ++it) {
            int id = lane + 32 * it;
            int el = id / 24, wc = id % 24, w = wc / 3, c = wc % 3;
            float s = 0.f;
#pragma unroll
            for (int v = 0; v < 8; ++v)
                s += sh1b[el * 24 + v * 3 + c] * gbuf[el * 64 + v * 8 + w];
            vacc[it] += s;
        }
    }

    // ---- block-level reduction ----
#pragma unroll
    for (int q = 0; q < 4; ++q) atomicAdd(&maccs[lane * 4 + q], macc[q]);
#pragma unroll
    for (int it = 0; it < 12; ++it) {
        int wc = (lane + 32 * it) % 24;
        atomicAdd(&vaccs[wc], vacc[it] * 0.03125f);   // / sqrt(128*8)
    }
    __syncthreads();

    if (tid < 128) out_mi[i * 128 + tid] = maccs[tid];
    if (tid < 24)  out_pos[i * 24 + tid] = pos[i * 24 + tid] + vaccs[tid] * (1.0f / 511.0f);
}

extern "C" void kernel_launch(void* const* d_in, const int* in_sizes, int n_in,
                              void* d_out, int out_size, void* d_ws, size_t ws_size,
                              hipStream_t stream) {
    const float* pos     = (const float*)d_in[0];
    const float* feat    = (const float*)d_in[1];
    const int*   senders = (const int*)  d_in[2];
    const float* We0     = (const float*)d_in[3];
    const float* We1     = (const float*)d_in[4];
    const float* winf    = (const float*)d_in[5];
    const float* Wx0     = (const float*)d_in[6];
    const float* Wx1     = (const float*)d_in[7];
    const float* Wtp     = (const float*)d_in[8];
    float* out_pos = (float*)d_out;                  // [512*8*3]
    float* out_mi  = (float*)d_out + NNODES * 24;    // [512*128]
    (void)in_sizes; (void)n_in; (void)out_size; (void)d_ws; (void)ws_size;
    egcl_fused<<<dim3(NNODES), dim3(256), SMEM_TOTAL, stream>>>(
        pos, feat, senders, We0, We1, winf, Wx0, Wx1, Wtp, out_pos, out_mi);
}
